// AdaptiveSpectrumLayer_26139170963693
// MI455X (gfx1250) — matrix operand order repaired and checked by reading, not compiled, not run on hardware
//
#include <hip/hip_runtime.h>
#include <math.h>

typedef __attribute__((ext_vector_type(16))) _Float16 v16h;
typedef __attribute__((ext_vector_type(8)))  _Float16 h8;
typedef __attribute__((ext_vector_type(8)))  float    v8f;

// Problem dims
constexpr int Bn  = 64;
constexpr int Hn  = 512;
constexpr int Fn  = 128;
constexpr int HID = 32;
constexpr int NF  = 257;            // H/2+1
constexpr int MP  = 272;            // NF padded to 16 (GEMM M)
constexpr int KNF = 288;            // NF padded to 32 (GEMM K, inverse)
constexpr int KG  = 1056;           // 4*NF=1028 padded to 32 (gate K)
constexpr int NG  = 272;            // NF padded to 16 (gate N)
constexpr float RSQ = 0.04419417382415922f;   // 1/sqrt(512)
constexpr float W512 = 0.012271846303085129f; // 2*pi/512

// Workspace layout (all 256B aligned)
constexpr size_t OFF_CCOS = 0;
constexpr size_t OFF_CSIN = OFF_CCOS + (size_t)MP * Hn * 2;        // 278528
constexpr size_t OFF_ICC  = OFF_CSIN + (size_t)MP * Hn * 2;
constexpr size_t OFF_ICS  = OFF_ICC  + (size_t)Hn * KNF * 2;       // 294912
constexpr size_t OFF_XHT  = OFF_ICS  + (size_t)Hn * KNF * 2;
constexpr size_t OFF_RE   = OFF_XHT  + (size_t)Bn * Fn * Hn * 2;   // 8388608
constexpr size_t OFF_IM   = OFF_RE   + (size_t)Bn * MP * Fn * 4;   // 8912896
constexpr size_t OFF_FFH  = OFF_IM   + (size_t)Bn * MP * Fn * 4;
constexpr size_t OFF_W2T  = OFF_FFH  + (size_t)Bn * Fn * KG * 2;   // 17301504
constexpr size_t OFF_BG2  = OFF_W2T  + (size_t)NG * KG * 2;        // 574464
constexpr size_t OFF_G    = OFF_BG2  + 1280;
constexpr size_t OFF_NRE  = OFF_G    + (size_t)Bn * Fn * NG * 4;   // 8912896
constexpr size_t OFF_NIM  = OFF_NRE  + (size_t)Bn * Fn * KNF * 4;  // 9437184
constexpr size_t WS_TOTAL = OFF_NIM  + (size_t)Bn * Fn * KNF * 4;  // ~73 MB

__device__ __forceinline__ float sigmoidf(float x) {
  return 1.0f / (1.0f + __expf(-x));
}

// ---------------- table builders ----------------
__global__ __launch_bounds__(256) void k_fwd_tables(_Float16* cc, _Float16* cs) {
  int tid = blockIdx.x * 256 + threadIdx.x;
  if (tid >= MP * Hn) return;
  int m = tid / Hn, t = tid % Hn;
  float c = 0.f, s = 0.f;
  if (m < NF) {
    int r = (m * t) & (Hn - 1);           // exact argument reduction
    float a = (float)r * W512;
    c =  cosf(a) * RSQ;
    s = -sinf(a) * RSQ;                    // Im = -sum x*sin
  }
  cc[tid] = (_Float16)c;
  cs[tid] = (_Float16)s;
}

__global__ __launch_bounds__(256) void k_inv_tables(_Float16* ic, _Float16* is) {
  int tid = blockIdx.x * 256 + threadIdx.x;
  if (tid >= Hn * KNF) return;
  int t = tid / KNF, n = tid % KNF;
  float c = 0.f, s = 0.f;
  if (n < NF) {
    float w = (n == 0 || n == NF - 1) ? 1.f : 2.f;  // Hermitian fold weights
    int r = (n * t) & (Hn - 1);
    float a = (float)r * W512;
    c =  w * cosf(a) * RSQ;
    s = -w * sinf(a) * RSQ;
  }
  ic[tid] = (_Float16)c;
  is[tid] = (_Float16)s;
}

// x (B,H,F) f32 -> xhT (B,F,H) f16
__global__ __launch_bounds__(256) void k_transpose_x(const float* __restrict__ x,
                                                     _Float16* __restrict__ xhT) {
  size_t tid = (size_t)blockIdx.x * 256 + threadIdx.x;
  if (tid >= (size_t)Bn * Hn * Fn) return;
  int b = (int)(tid / (Hn * Fn));
  int rem = (int)(tid % (Hn * Fn));
  int t = rem / Fn, f = rem % Fn;
  xhT[(size_t)b * Fn * Hn + (size_t)f * Hn + t] = (_Float16)x[tid];
}

// W2T[j][n*4+k] = sum_h Wp[n,k,h] * Wg[n*32+h, j]   (transposed, f16)
__global__ __launch_bounds__(256) void k_w2(const float* __restrict__ Wp,
                                            const float* __restrict__ Wg,
                                            _Float16* __restrict__ W2T) {
  int tid = blockIdx.x * 256 + threadIdx.x;
  if (tid >= 4 * NF * NF) return;
  int i = tid / NF;          // 0..1027 = n*4+k
  int j = tid % NF;
  int n = i >> 2, k = i & 3;
  const float* wp = Wp + (size_t)(n * 4 + k) * HID;
  float s = 0.f;
  #pragma unroll
  for (int h = 0; h < HID; ++h)
    s += wp[h] * Wg[(size_t)(n * HID + h) * NF + j];
  W2T[(size_t)j * KG + i] = (_Float16)s;
}

// bg2[j] = bg[j] + sum_i bp[i]*Wg[i,j]
__global__ __launch_bounds__(256) void k_bg2(const float* __restrict__ bp,
                                             const float* __restrict__ Wg,
                                             const float* __restrict__ bg,
                                             float* __restrict__ bg2) {
  __shared__ float red[256];
  int j = blockIdx.x;
  float s = 0.f;
  for (int i = threadIdx.x; i < NF * HID; i += 256)
    s += bp[i] * Wg[(size_t)i * NF + j];
  red[threadIdx.x] = s;
  __syncthreads();
  for (int k = 128; k > 0; k >>= 1) {
    if (threadIdx.x < k) red[threadIdx.x] += red[threadIdx.x + k];
    __syncthreads();
  }
  if (threadIdx.x == 0) bg2[j] = bg[j] + red[0];
}

// ---------------- WMMA fragment helpers ----------------
// A fragment, row-major A (lda halves): 16x32 f16, K contiguous (k = 16*hi + s)
__device__ __forceinline__ v16h load_afrag(const _Float16* __restrict__ A,
                                           int lda, int row, int k0, int hi) {
  return *(const v16h*)(A + (size_t)row * lda + k0 + hi * 16);
}
// B fragment from transposed B (N-major, ldb halves): K contiguous
__device__ __forceinline__ v16h load_bfrag(const _Float16* __restrict__ Bt,
                                           int ldb, int col, int k0, int hi) {
  return *(const v16h*)(Bt + (size_t)col * ldb + k0 + hi * 16);
}

// ---------------- forward DFT: Re/Im = [Ccos;Csin] @ x_b ----------------
__global__ __launch_bounds__(256) void k_dft_fwd(const _Float16* __restrict__ Ccos,
                                                 const _Float16* __restrict__ Csin,
                                                 const _Float16* __restrict__ xhT,
                                                 float* __restrict__ Re,
                                                 float* __restrict__ Im) {
  int wid = blockIdx.x * 8 + (threadIdx.x >> 5);
  int lane = threadIdx.x & 31;
  int b = wid / ((MP / 16) * (Fn / 16));
  int r = wid % ((MP / 16) * (Fn / 16));
  int mt = r >> 3, nt = r & 7;
  int row = mt * 16 + (lane & 15);
  int col = nt * 16 + (lane & 15);
  int hi = lane >> 4;
  const _Float16* bt = xhT + (size_t)b * Fn * Hn;
  v8f aRe = {}, aIm = {};
  #pragma unroll
  for (int kt = 0; kt < Hn / 32; ++kt) {
    int k0 = kt * 32;
    v16h bf = load_bfrag(bt, Hn, col, k0, hi);
    v16h ac = load_afrag(Ccos, Hn, row, k0, hi);
    v16h as = load_afrag(Csin, Hn, row, k0, hi);
    aRe = __builtin_amdgcn_wmma_f32_16x16x32_f16(false, ac, false, bf, (short)0, aRe, false, false);
    aIm = __builtin_amdgcn_wmma_f32_16x16x32_f16(false, as, false, bf, (short)0, aIm, false, false);
  }
  size_t base = (size_t)b * MP * Fn;
  #pragma unroll
  for (int i = 0; i < 8; ++i) {
    int m = mt * 16 + hi * 8 + i;
    Re[base + (size_t)m * Fn + col] = aRe[i];
    Im[base + (size_t)m * Fn + col] = aIm[i];
  }
}

// ---------------- features + per-freq proj + complexifier ----------------
__global__ __launch_bounds__(256) void k_features(const float* __restrict__ Re,
                                                  const float* __restrict__ Im,
                                                  const float* __restrict__ Wp,
                                                  const float* __restrict__ bp,
                                                  const float* __restrict__ Wm,
                                                  const float* __restrict__ bm,
                                                  const float* __restrict__ Wph,
                                                  const float* __restrict__ bph,
                                                  _Float16* __restrict__ ffh,
                                                  float* __restrict__ nReT,
                                                  float* __restrict__ nImT) {
  int tid = blockIdx.x * 256 + threadIdx.x;
  if (tid >= NF * Fn) return;
  int b = blockIdx.y;
  int n = tid / Fn, f = tid % Fn;
  float re = Re[(size_t)b * MP * Fn + (size_t)n * Fn + f];
  float im = Im[(size_t)b * MP * Fn + (size_t)n * Fn + f];
  float mag = sqrtf(re * re + im * im);
  float inv = (mag > 0.f) ? 1.f / mag : 0.f;
  float sn = im * inv;
  float cs = (mag > 0.f) ? re * inv : 1.f;
  float fr = (float)n * (100.0f / 512.0f);
  _Float16* fp = ffh + (size_t)(b * Fn + f) * KG + n * 4;
  fp[0] = (_Float16)mag; fp[1] = (_Float16)sn;
  fp[2] = (_Float16)cs;  fp[3] = (_Float16)fr;
  const float* wp = Wp + (size_t)n * 4 * HID;
  const float* bn = bp + (size_t)n * HID;
  const float* wm = Wm + (size_t)n * HID;
  const float* wq = Wph + (size_t)n * HID;
  float md = 0.f, pd = 0.f;
  #pragma unroll
  for (int h = 0; h < HID; ++h) {
    float p = fmaf(mag, wp[h], fmaf(sn, wp[HID + h],
              fmaf(cs, wp[2 * HID + h], fmaf(fr, wp[3 * HID + h], bn[h]))));
    float xr = fmaxf(p, 0.f);
    md = fmaf(xr, wm[h], md);
    pd = fmaf(xr, wq[h], pd);
  }
  float m = fmaxf(md + bm[n], 0.f);
  float ph = 6.283185307179586f * sigmoidf(pd + bph[n]);
  float sp, cp;
  __sincosf(ph, &sp, &cp);
  size_t t = (size_t)b * Fn * KNF + (size_t)f * KNF + n;
  nReT[t] = m * cp;
  nImT[t] = m * sp;
}

// ---------------- gate GEMM: g = ffh @ W2 (K folded, N-major B) ----------------
__global__ __launch_bounds__(256) void k_gate(const _Float16* __restrict__ ffh,
                                              const _Float16* __restrict__ W2T,
                                              float* __restrict__ g) {
  int wid = blockIdx.x * 8 + (threadIdx.x >> 5);
  int lane = threadIdx.x & 31;
  int mt = wid / (NG / 16);
  int nt = wid % (NG / 16);
  int row = mt * 16 + (lane & 15);
  int col = nt * 16 + (lane & 15);
  int hi = lane >> 4;
  v8f acc = {};
  for (int kt = 0; kt < KG / 32; ++kt) {
    int k0 = kt * 32;
    v16h af = load_afrag(ffh, KG, row, k0, hi);
    v16h bf = load_bfrag(W2T, KG, col, k0, hi);
    acc = __builtin_amdgcn_wmma_f32_16x16x32_f16(false, af, false, bf, (short)0, acc, false, false);
  }
  #pragma unroll
  for (int i = 0; i < 8; ++i) {
    int m = mt * 16 + hi * 8 + i;
    g[(size_t)m * NG + col] = acc[i];
  }
}

// ---------------- blend: w = sigmoid(silu(g+bg2)); spec = w*new + (1-w)*orig ----------------
__global__ __launch_bounds__(256) void k_blend(const float* __restrict__ g,
                                               const float* __restrict__ bg2,
                                               const float* __restrict__ Re,
                                               const float* __restrict__ Im,
                                               float* __restrict__ nReT,
                                               float* __restrict__ nImT) {
  int tid = blockIdx.x * 256 + threadIdx.x;
  if (tid >= NF * Fn) return;
  int b = blockIdx.y;
  int n = tid / Fn, f = tid % Fn;
  float gv = g[(size_t)(b * Fn + f) * NG + n] + bg2[n];
  float gs = gv * sigmoidf(gv);
  float w = sigmoidf(gs);
  size_t t = (size_t)b * Fn * KNF + (size_t)f * KNF + n;
  float oRe = Re[(size_t)b * MP * Fn + (size_t)n * Fn + f];
  float oIm = Im[(size_t)b * MP * Fn + (size_t)n * Fn + f];
  nReT[t] = fmaf(w, nReT[t] - oRe, oRe);
  nImT[t] = fmaf(w, nImT[t] - oIm, oIm);
}

// ---------------- spec f32 -> f16 for inverse GEMM ----------------
__global__ __launch_bounds__(256) void k_spec_h(const float* __restrict__ nReT,
                                                const float* __restrict__ nImT,
                                                _Float16* __restrict__ sReT,
                                                _Float16* __restrict__ sImT) {
  size_t tid = (size_t)blockIdx.x * 256 + threadIdx.x;
  if (tid >= (size_t)Bn * Fn * KNF) return;
  sReT[tid] = (_Float16)nReT[tid];
  sImT[tid] = (_Float16)nImT[tid];
}

// ---------------- inverse DFT: out = ICcos@WRe + ICsin@WIm ----------------
__global__ __launch_bounds__(256) void k_dft_inv(const _Float16* __restrict__ ICc,
                                                 const _Float16* __restrict__ ICs,
                                                 const _Float16* __restrict__ sReT,
                                                 const _Float16* __restrict__ sImT,
                                                 float* __restrict__ out) {
  int wid = blockIdx.x * 8 + (threadIdx.x >> 5);
  int lane = threadIdx.x & 31;
  int b = wid / ((Hn / 16) * (Fn / 16));
  int r = wid % ((Hn / 16) * (Fn / 16));
  int mt = r >> 3, nt = r & 7;
  int row = mt * 16 + (lane & 15);
  int col = nt * 16 + (lane & 15);
  int hi = lane >> 4;
  const _Float16* br = sReT + (size_t)b * Fn * KNF;
  const _Float16* bi = sImT + (size_t)b * Fn * KNF;
  v8f acc = {};
  #pragma unroll
  for (int kt = 0; kt < KNF / 32; ++kt) {
    int k0 = kt * 32;
    v16h ac = load_afrag(ICc, KNF, row, k0, hi);
    v16h as = load_afrag(ICs, KNF, row, k0, hi);
    v16h bfr = load_bfrag(br, KNF, col, k0, hi);
    v16h bfi = load_bfrag(bi, KNF, col, k0, hi);
    acc = __builtin_amdgcn_wmma_f32_16x16x32_f16(false, ac, false, bfr, (short)0, acc, false, false);
    acc = __builtin_amdgcn_wmma_f32_16x16x32_f16(false, as, false, bfi, (short)0, acc, false, false);
  }
  size_t base = (size_t)b * Hn * Fn;
  #pragma unroll
  for (int i = 0; i < 8; ++i) {
    int m = mt * 16 + hi * 8 + i;
    out[base + (size_t)m * Fn + col] = acc[i];
  }
}

extern "C" void kernel_launch(void* const* d_in, const int* in_sizes, int n_in,
                              void* d_out, int out_size, void* d_ws, size_t ws_size,
                              hipStream_t stream) {
  (void)in_sizes; (void)n_in; (void)out_size;
  const float* x   = (const float*)d_in[0];
  const float* Wp  = (const float*)d_in[1];
  const float* bp  = (const float*)d_in[2];
  const float* Wg  = (const float*)d_in[3];
  const float* bg  = (const float*)d_in[4];
  const float* Wm  = (const float*)d_in[5];
  const float* bm  = (const float*)d_in[6];
  const float* Wph = (const float*)d_in[7];
  const float* bph = (const float*)d_in[8];
  float* out = (float*)d_out;

  char* ws = (char*)d_ws;
  _Float16* Ccos = (_Float16*)(ws + OFF_CCOS);
  _Float16* Csin = (_Float16*)(ws + OFF_CSIN);
  _Float16* ICc  = (_Float16*)(ws + OFF_ICC);
  _Float16* ICs  = (_Float16*)(ws + OFF_ICS);
  _Float16* xhT  = (_Float16*)(ws + OFF_XHT);
  float*    Re   = (float*)   (ws + OFF_RE);
  float*    Im   = (float*)   (ws + OFF_IM);
  _Float16* ffh  = (_Float16*)(ws + OFF_FFH);
  _Float16* W2T  = (_Float16*)(ws + OFF_W2T);
  float*    bg2  = (float*)   (ws + OFF_BG2);
  float*    g    = (float*)   (ws + OFF_G);
  float*    nReT = (float*)   (ws + OFF_NRE);
  float*    nImT = (float*)   (ws + OFF_NIM);
  // reuse f32 Re/Im buffers' tail? no — convert spec into ffh region? keep separate:
  // sReT/sImT reuse the (now consumed) xhT buffer: B*F*KNF halves = 4.7MB each <= 8.4MB? 2x needed.
  _Float16* sReT = (_Float16*)(ws + OFF_XHT);                       // reuse xhT (8.4MB)
  _Float16* sImT = (_Float16*)(ws + OFF_FFH);                       // reuse ffh (17.3MB)

  size_t zbytes = (ws_size < WS_TOTAL) ? ws_size : WS_TOTAL;
  hipMemsetAsync(d_ws, 0, zbytes, stream);

  k_fwd_tables<<<(MP * Hn + 255) / 256, 256, 0, stream>>>(Ccos, Csin);
  k_inv_tables<<<(Hn * KNF + 255) / 256, 256, 0, stream>>>(ICc, ICs);
  k_transpose_x<<<(Bn * Hn * Fn + 255) / 256, 256, 0, stream>>>(x, xhT);
  k_w2<<<(4 * NF * NF + 255) / 256, 256, 0, stream>>>(Wp, Wg, W2T);
  k_bg2<<<NF, 256, 0, stream>>>(bp, Wg, bg, bg2);

  k_dft_fwd<<<(Bn * (MP / 16) * (Fn / 16)) / 8, 256, 0, stream>>>(Ccos, Csin, xhT, Re, Im);

  dim3 gf((NF * Fn + 255) / 256, Bn);
  k_features<<<gf, 256, 0, stream>>>(Re, Im, Wp, bp, Wm, bm, Wph, bph, ffh, nReT, nImT);

  k_gate<<<((Bn * Fn / 16) * (NG / 16)) / 8, 256, 0, stream>>>(ffh, W2T, g);

  k_blend<<<gf, 256, 0, stream>>>(g, bg2, Re, Im, nReT, nImT);

  k_spec_h<<<((size_t)Bn * Fn * KNF + 255) / 256, 256, 0, stream>>>(nReT, nImT, sReT, sImT);

  k_dft_inv<<<(Bn * (Hn / 16) * (Fn / 16)) / 8, 256, 0, stream>>>(ICc, ICs, sReT, sImT, out);
}
